// LSTM_67637144977530
// MI455X (gfx1250) — matrix operand order repaired and checked by reading, not compiled, not run on hardware
//
#include <hip/hip_runtime.h>
#include <stdint.h>

// ---------------------------------------------------------------------------
// LSTM: B=8192, T=28, I=28, H=128 (gates 4H=512), C=10, fp32 in/out.
// f16 WMMA (16x16x32) with f32 accumulation; weights stationary in VGPRs;
// h-state recirculated through 4KB of LDS; x tile staged via CDNA5 async
// global->LDS copies; native v_tanh_f32 for gate activations.
// One workgroup (8 waves) = 16 batch rows.
// ---------------------------------------------------------------------------

#define BATCH 8192
#define TSTEPS 28
#define IDIM 28
#define HDIM 128
#define GDIM 512   // 4*H

typedef _Float16 v16h __attribute__((ext_vector_type(16)));
typedef _Float16 v8h  __attribute__((ext_vector_type(8)));
typedef float    v8f  __attribute__((ext_vector_type(8)));

__device__ __forceinline__ v16h cat16(v8h a, v8h b) {
  return __builtin_shufflevector(a, b, 0,1,2,3,4,5,6,7,8,9,10,11,12,13,14,15);
}
// B-fragment: 16 contiguous halves (one 32B run), 32B-aligned.
__device__ __forceinline__ v16h load_g16(const _Float16* __restrict__ p) {
  v8h lo = *(const v8h*)p;
  v8h hi = *(const v8h*)(p + 8);
  return cat16(lo, hi);
}
// A-fragment: two 16B runs from LDS.
__device__ __forceinline__ v16h load_s16(const _Float16* p0, const _Float16* p1) {
  v8h lo = *(const v8h*)p0;
  v8h hi = *(const v8h*)p1;
  return cat16(lo, hi);
}

// Native gfx1250 tanh (single TRANS op); branch-free activations.
#if __has_builtin(__builtin_amdgcn_tanhf)
__device__ __forceinline__ float ftanh(float x) { return __builtin_amdgcn_tanhf(x); }
#elif __has_builtin(__builtin_amdgcn_tanh_f32)
__device__ __forceinline__ float ftanh(float x) { return __builtin_amdgcn_tanh_f32(x); }
#else
__device__ __forceinline__ float ftanh(float x) { return tanhf(x); }
#endif
__device__ __forceinline__ float fsig(float x) {
  return 0.5f * ftanh(0.5f * x) + 0.5f;
}

// ---------------------------------------------------------------------------
// Prep: fp32 weights -> f16 (W_ih zero-padded K 28->32), fuse biases.
// ---------------------------------------------------------------------------
__global__ void lstm_prep(const float* __restrict__ Wih, const float* __restrict__ Whh,
                          const float* __restrict__ bih, const float* __restrict__ bhh,
                          _Float16* __restrict__ wih_p, _Float16* __restrict__ whh_p,
                          float* __restrict__ bias) {
  int idx = blockIdx.x * blockDim.x + threadIdx.x;
  if (idx < GDIM * 32) {
    int g = idx >> 5, k = idx & 31;
    float v = (k < IDIM) ? Wih[g * IDIM + k] : 0.0f;
    wih_p[idx] = (_Float16)v;
  }
  if (idx < GDIM * HDIM) whh_p[idx] = (_Float16)Whh[idx];
  if (idx < GDIM) bias[idx] = bih[idx] + bhh[idx];
}

// ---------------------------------------------------------------------------
// Main fused LSTM kernel. Grid: 512 blocks x 256 threads (8 waves).
// ---------------------------------------------------------------------------
__global__ __launch_bounds__(256) void lstm_main(
    const float*    __restrict__ x,     // [B, T, I]
    const _Float16* __restrict__ wih,   // [512, 32] padded f16
    const _Float16* __restrict__ whh,   // [512, 128] f16
    const float*    __restrict__ bias,  // [512]
    const float*    __restrict__ wfc,   // [10, 128]
    const float*    __restrict__ bfc,   // [10]
    float*          __restrict__ out)   // [B, 10]
{
  __shared__ float    stage[16 * TSTEPS * IDIM];   // raw x tile (fp32), 50176 B
  __shared__ _Float16 xt[TSTEPS][16][32];          // padded f16 x tile, 28672 B
  __shared__ _Float16 hl[16][HDIM];                // h state, 4096 B

  const int tid = threadIdx.x;
  const int b0  = blockIdx.x * 16;

  // ---- Stage x tile: 16 rows x 784 floats, contiguous per row -> async LDS ----
  {
    uint64_t gbase = (uint64_t)(uintptr_t)(x + (size_t)b0 * (TSTEPS * IDIM));
    uint32_t lds0  = (uint32_t)(uintptr_t)(&stage[0]);
    const int nvec = (16 * TSTEPS * IDIM) / 4;     // 3136 b128 transfers
    for (int i = tid; i < nvec; i += 256) {
      uint32_t voff = (uint32_t)i * 16u;
      uint32_t ldst = lds0 + voff;
      asm volatile("global_load_async_to_lds_b128 %0, %1, %2"
                   :: "v"(ldst), "v"(voff), "s"(gbase) : "memory");
    }
    asm volatile("s_wait_asynccnt 0" ::: "memory");
  }
  __syncthreads();

  // ---- Convert to zero-padded f16, WMMA-friendly layout [t][m][k<32] ----
  for (int e = tid; e < TSTEPS * 16 * 32; e += 256) {
    int t = e >> 9;           // / 512
    int r = e & 511;
    int m = r >> 5;
    int k = r & 31;
    float v = (k < IDIM) ? stage[m * (TSTEPS * IDIM) + t * IDIM + k] : 0.0f;
    xt[t][m][k] = (_Float16)v;
  }
  __syncthreads();

  // ---- Per-wave setup: wave w owns h columns [16w, 16w+16) ----
  const int lane = tid & 31;
  const int w    = tid >> 5;
  const int n    = lane & 15;       // output column within tile / A row
  const int hi   = lane >> 4;       // lane half
  const int wcol = w * 16;
  const int hi8  = hi * 8;
  const int m    = n;               // A-matrix row for this lane

  // Pin B-fragments (weights) in VGPRs for the whole recurrence.
  v16h bihf[4];
  v16h bhhf[4][4];
  float bb[4];
#pragma unroll
  for (int g = 0; g < 4; ++g) {
    int col = g * HDIM + wcol + n;
    bihf[g] = load_g16(wih + (size_t)col * 32 + hi * 16);
#pragma unroll
    for (int kc = 0; kc < 4; ++kc)
      bhhf[g][kc] = load_g16(whh + (size_t)col * HDIM + kc * 32 + hi * 16);
    bb[g] = bias[col];
  }

  v8f c = {};   // cell state tile, resident in VGPRs

  for (int t = 0; t < TSTEPS; ++t) {
    // gates start at zero (bias folded into elementwise phase).
    v8f acc[4];
#pragma unroll
    for (int g = 0; g < 4; ++g) acc[g] = (v8f){};

    // gates += x_t @ W_ih^T   (K=32, one WMMA per gate)
    {
      v16h ax = load_s16(&xt[t][m][hi * 16], &xt[t][m][hi * 16 + 8]);
#pragma unroll
      for (int g = 0; g < 4; ++g)
        acc[g] = __builtin_amdgcn_wmma_f32_16x16x32_f16(
            false, ax, false, bihf[g], (short)0, acc[g], false, false);
    }

    // gates += h @ W_hh^T   (K=128 in 4 chunks x 4 gates = 16 WMMAs)
    if (t > 0) {
#pragma unroll
      for (int kc = 0; kc < 4; ++kc) {
        v16h ah = load_s16(&hl[m][kc * 32 + hi * 16], &hl[m][kc * 32 + hi * 16 + 8]);
#pragma unroll
        for (int g = 0; g < 4; ++g)
          acc[g] = __builtin_amdgcn_wmma_f32_16x16x32_f16(
              false, ah, false, bhhf[g][kc], (short)0, acc[g], false, false);
      }
    }

    __syncthreads();   // all waves done reading previous h

    // Elementwise cell update (branch-free, native v_tanh); write h to LDS.
#pragma unroll
    for (int j = 0; j < 8; ++j) {
      float ig = fsig(acc[0][j] + bb[0]);
      float fg = fsig(acc[1][j] + bb[1]);
      float gg = ftanh(acc[2][j] + bb[2]);
      float og = fsig(acc[3][j] + bb[3]);
      float cv = fg * c[j] + ig * gg;
      c[j] = cv;
      float hv = og * ftanh(cv);
      hl[hi8 + j][wcol + n] = (_Float16)hv;   // row M = hi*8+j, col = wcol+n
    }

    __syncthreads();   // new h visible to all waves
  }

  // ---- Final FC (128 -> 10) + ReLU, tiny: one thread per (row, class) ----
  if (tid < 160) {
    int row = tid / 10, cls = tid % 10;
    float a = bfc[cls];
#pragma unroll 8
    for (int k = 0; k < HDIM; ++k)
      a = fmaf((float)hl[row][k], wfc[cls * HDIM + k], a);
    out[(b0 + row) * 10 + cls] = fmaxf(a, 0.0f);
  }
}

// ---------------------------------------------------------------------------
extern "C" void kernel_launch(void* const* d_in, const int* in_sizes, int n_in,
                              void* d_out, int out_size, void* d_ws, size_t ws_size,
                              hipStream_t stream) {
  const float* x   = (const float*)d_in[0];
  const float* Wih = (const float*)d_in[1];
  const float* Whh = (const float*)d_in[2];
  const float* bih = (const float*)d_in[3];
  const float* bhh = (const float*)d_in[4];
  const float* Wfc = (const float*)d_in[5];
  const float* bfc = (const float*)d_in[6];
  float* out = (float*)d_out;

  char* ws = (char*)d_ws;
  _Float16* wih_p = (_Float16*)(ws);                    // 512*32*2  = 32768 B
  _Float16* whh_p = (_Float16*)(ws + 32768);            // 512*128*2 = 131072 B
  float*    bias  = (float*)(ws + 32768 + 131072);      // 512*4     = 2048 B

  lstm_prep<<<256, 256, 0, stream>>>(Wih, Whh, bih, bhh, wih_p, whh_p, bias);
  lstm_main<<<BATCH / 16, 256, 0, stream>>>(x, wih_p, whh_p, bias, Wfc, bfc, out);
}
